// GlobalAttention_9208409882894
// MI455X (gfx1250) — matrix operand order repaired and checked by reading, not compiled, not run on hardware
//
#include <hip/hip_runtime.h>

typedef __bf16 bf16;
typedef __attribute__((ext_vector_type(16))) __bf16 v16bf;
typedef __attribute__((ext_vector_type(8)))  __bf16 v8bf;
typedef __attribute__((ext_vector_type(8)))  float  v8f;

#define NB   128
#define NN   2048
#define NC   256
#define NH   8
#define NKD  32
#define NOUT 256
#define LDA  264   // padded bf16 row stride (33*16B) -> conflict-free ds_read_b128

// ---------------- WMMA helpers ----------------

__device__ __forceinline__ v8f wmma_bf16(v16bf a, v16bf b, v8f c) {
  return __builtin_amdgcn_wmma_f32_16x16x32_bf16(false, a, false, b, (short)0, c,
                                                 false, false);
}

__device__ __forceinline__ v16bf cat16(v8bf lo, v8bf hi) {
  return __builtin_shufflevector(lo, hi, 0,1,2,3,4,5,6,7,8,9,10,11,12,13,14,15);
}

// A-fragment (16x32 bf16) from LDS row-major tile, per CDNA5 ISA layout:
// lanes 0-15 hold K[k0..k0+7] and K[k0+16..k0+23] for row M=lane,
// lanes 16-31 hold the +8 halves for row M=lane-16.
__device__ __forceinline__ v16bf load_afrag(const bf16* As, int strip, int lane, int k0) {
  int row = strip*16 + (lane & 15);
  int kb  = k0 + ((lane & 16) ? 8 : 0);
  const bf16* p = As + row*LDA + kb;
  v8bf a0 = *(const v8bf*)p;
  v8bf a1 = *(const v8bf*)(p + 16);
  return cat16(a0, a1);
}

// B-fragment (32x16 bf16) from pre-packed global weights: frag = 32 lanes x 16 bf16.
__device__ __forceinline__ v16bf load_bfrag(const bf16* wp, int frag, int lane) {
  const bf16* p = wp + (size_t)frag*512 + (size_t)lane*16;
  v8bf b0 = *(const v8bf*)p;
  v8bf b1 = *(const v8bf*)(p + 8);
  return cat16(b0, b1);
}

// Stage a 64x256 fp32 tile -> bf16 LDS tile (coalesced float4 global reads).
__device__ __forceinline__ void stage_tile(const float* __restrict__ src, bf16* As, int t) {
  const float4* s4 = (const float4*)src;
  #pragma unroll 4
  for (int i = t; i < 64*64; i += 256) {
    float4 v = s4[i];
    bf16* d = As + (i >> 6)*LDA + ((i & 63) << 2);
    d[0] = (bf16)v.x; d[1] = (bf16)v.y; d[2] = (bf16)v.z; d[3] = (bf16)v.w;
  }
}

// ---------------- K0: pack weights into bf16 B-fragment order ----------------
// Fragment (kt, nt): lanes 0-15 -> col nt*16+lane, K = kt*32 + {0..7, 16..23}
//                    lanes 16-31 -> col nt*16+lane-16, K = kt*32 + {8..15, 24..31}
__global__ void pack_wmma_b(const float* __restrict__ gw, const float* __restrict__ ow,
                            const float* __restrict__ kw, const float* __restrict__ vw,
                            bf16* __restrict__ wgp, bf16* __restrict__ wop,
                            bf16* __restrict__ wkvp) {
  int frag = blockIdx.x;
  int lane = threadIdx.x;
  const float* src; bf16* dst; int nt_count;
  if (frag < 128)      { src = gw; dst = wgp;  nt_count = 16; }
  else if (frag < 256) { src = ow; dst = wop;  nt_count = 16; frag -= 128; }
  else                 { src = nullptr; dst = wkvp; nt_count = 4; frag -= 256; }
  int col = (frag % nt_count)*16 + (lane & 15);
  int kb  = (frag / nt_count)*32 + ((lane & 16) ? 8 : 0);
  bf16* out = dst + (size_t)frag*512 + (size_t)lane*16;
  for (int j = 0; j < 16; ++j) {
    float w;
    int k = kb + (j < 8 ? j : j + 8);
    if (src) w = src[(size_t)k*256 + col];
    else     w = (col < NKD) ? kw[(size_t)k*NKD + col]
                             : vw[(size_t)k*NKD + (col - NKD)];
    out[j] = (bf16)w;
  }
}

// ---------------- K1: masked mean pool over N ----------------
__global__ void pool_query(const float* __restrict__ qd, const float* __restrict__ qm,
                           float* __restrict__ qavg) {
  int b = blockIdx.x, t = threadIdx.x;       // 256 threads = channels
  const float* qb = qd + (size_t)b*NN*NC;
  const float* mb = qm + (size_t)b*NN;
  float s = 0.f, ms = 0.f;
  for (int n = 0; n < NN; ++n) {
    float m = mb[n];
    s  += m * qb[(size_t)n*NC + t];
    ms += m;
  }
  qavg[b*NC + t] = s / (ms + 1e-10f);
}

// ---------------- K2: q = q_avg @ query_w * KD^-0.5 ----------------
__global__ void project_q(const float* __restrict__ qavg, const float* __restrict__ qw,
                          float* __restrict__ q) {
  int b = blockIdx.x, t = threadIdx.x;
  __shared__ float qa[NC];
  qa[t] = qavg[b*NC + t];
  __syncthreads();
  float acc = 0.f;
  for (int c = 0; c < NC; ++c) acc += qa[c] * qw[(size_t)c*256 + t];
  q[b*256 + t] = acc * 0.17677669529663687f;   // 32^-0.5
}

// ---------------- K3: k/v projection GEMM [R,256]x[256,64] (WMMA bf16) ----------
__global__ void kv_proj_kernel(const float* __restrict__ mdat,
                               const bf16* __restrict__ wkvp,
                               float* __restrict__ kbuf, float* __restrict__ vbuf) {
  extern __shared__ char smem[];
  bf16* As = (bf16*)smem;                       // 64 x LDA bf16
  int t = threadIdx.x, lane = t & 31, w = t >> 5;
  int strip = w >> 1, half = w & 1;             // 4 strips x 2 halves
  size_t r0 = (size_t)blockIdx.x * 64;
  stage_tile(mdat + r0*NC, As, t);
  __syncthreads();
  v8f zero = {0.f,0.f,0.f,0.f,0.f,0.f,0.f,0.f};
  v8f acc[2]; acc[0] = zero; acc[1] = zero;
  for (int kt = 0; kt < 8; ++kt) {
    v16bf a = load_afrag(As, strip, lane, kt*32);
    #pragma unroll
    for (int i = 0; i < 2; ++i) {
      v16bf bf = load_bfrag(wkvp, kt*4 + half*2 + i, lane);
      acc[i] = wmma_bf16(a, bf, acc[i]);
    }
  }
  int mhi = (lane & 16) ? 8 : 0;
  #pragma unroll
  for (int i = 0; i < 2; ++i) {
    int col = (half*2 + i)*16 + (lane & 15);
    float* dst; int c2;
    if (col < NKD) { dst = kbuf; c2 = col; } else { dst = vbuf; c2 = col - NKD; }
    #pragma unroll
    for (int r = 0; r < 8; ++r) {
      size_t rg = r0 + strip*16 + r + mhi;
      dst[rg*NKD + c2] = acc[i][r];
    }
  }
}

// ---------------- K5: attention pooling (one wave per head, online softmax) -----
__global__ void attention_pool(const float* __restrict__ kbuf, const float* __restrict__ vbuf,
                               const float* __restrict__ qh, const float* __restrict__ qm,
                               float* __restrict__ wa) {
  extern __shared__ char smem[];
  float* ks = (float*)smem;            // 256*33
  float* vs = ks + 256*33;             // 256*33
  float* ps = vs + 256*33;             // 8*256
  float* qs = ps + 8*256;              // 256
  int t = threadIdx.x, h = t >> 5, lane = t & 31;
  int b = blockIdx.x;
  qs[t] = qh[b*256 + t];
  const float* kb_ = kbuf + (size_t)b*NN*NKD;
  const float* vb_ = vbuf + (size_t)b*NN*NKD;
  const float* mb  = qm   + (size_t)b*NN;
  float m_run = -1e30f, l_run = 0.f, acc = 0.f;
  for (int tile = 0; tile < NN/256; ++tile) {
    __syncthreads();
    for (int i = t; i < 256*NKD; i += 256) {   // coalesced tile load, padded LDS
      int nl = i >> 5, c = i & 31;
      ks[nl*33 + c] = kb_[(size_t)tile*256*NKD + i];
      vs[nl*33 + c] = vb_[(size_t)tile*256*NKD + i];
    }
    __syncthreads();
    float lg[8]; float mx = -1e30f;
    #pragma unroll
    for (int i = 0; i < 8; ++i) {
      int nl = lane + i*32;
      float s = 0.f;
      #pragma unroll
      for (int j = 0; j < NKD; ++j) s += qs[h*32 + j] * ks[nl*33 + j];
      s += 1e9f * (mb[tile*256 + nl] - 1.0f);
      lg[i] = s; mx = fmaxf(mx, s);
    }
    for (int off = 16; off > 0; off >>= 1) mx = fmaxf(mx, __shfl_xor(mx, off, 32));
    float m_new = fmaxf(m_run, mx);
    float corr = __expf(m_run - m_new);
    float s_t = 0.f;
    #pragma unroll
    for (int i = 0; i < 8; ++i) {
      float p = __expf(lg[i] - m_new);
      ps[h*256 + lane + i*32] = p;
      s_t += p;
    }
    for (int off = 16; off > 0; off >>= 1) s_t += __shfl_xor(s_t, off, 32);
    l_run = l_run * corr + s_t;
    acc  *= corr;
    m_run = m_new;
    __syncthreads();
    #pragma unroll 8
    for (int nl = 0; nl < 256; ++nl)        // lane = value channel
      acc += ps[h*256 + nl] * vs[nl*33 + lane];
  }
  wa[b*256 + h*32 + lane] = acc / l_run;
}

// ---------------- K4: fused gate GEMM -> sigmoid*scale -> output GEMM ----------
__global__ void gate_output_kernel(const float* __restrict__ qdat,
                                   const bf16* __restrict__ wgp,
                                   const bf16* __restrict__ wop,
                                   const float* __restrict__ gb,
                                   const float* __restrict__ wa,
                                   const float* __restrict__ ob,
                                   float* __restrict__ out) {
  extern __shared__ char smem[];
  bf16* As = (bf16*)smem;                    // 64 x LDA bf16
  bf16* Gs = (bf16*)(smem + (size_t)64*LDA*sizeof(bf16));
  int t = threadIdx.x, lane = t & 31, w = t >> 5;
  int strip = w >> 1, half = w & 1;
  size_t r0 = (size_t)blockIdx.x * 64;
  int b = blockIdx.x >> 5;                   // 32 row-tiles per batch
  stage_tile(qdat + r0*NC, As, t);
  __syncthreads();
  v8f zero = {0.f,0.f,0.f,0.f,0.f,0.f,0.f,0.f};
  v8f acc[8];
  #pragma unroll
  for (int i = 0; i < 8; ++i) acc[i] = zero;
  // GEMM1: gate logits = q_data @ gating_w
  for (int kt = 0; kt < 8; ++kt) {
    v16bf a = load_afrag(As, strip, lane, kt*32);
    #pragma unroll
    for (int i = 0; i < 8; ++i) {
      v16bf bf = load_bfrag(wgp, kt*16 + half*8 + i, lane);
      acc[i] = wmma_bf16(a, bf, acc[i]);
    }
  }
  int mhi = (lane & 16) ? 8 : 0;
  // epilogue1: sigmoid(x + gating_b) * weighted_avg[col] -> bf16 tile in LDS
  #pragma unroll
  for (int i = 0; i < 8; ++i) {
    int col = (half*8 + i)*16 + (lane & 15);
    float gbc = gb[col];
    float wvc = wa[b*256 + col];
    #pragma unroll
    for (int r = 0; r < 8; ++r) {
      int row = strip*16 + r + mhi;
      float g = acc[i][r] + gbc;
      g = wvc / (1.0f + __expf(-g));
      Gs[row*LDA + col] = (bf16)g;
    }
  }
  __syncthreads();
  #pragma unroll
  for (int i = 0; i < 8; ++i) acc[i] = zero;
  // GEMM2: out = gated @ output_w
  for (int kt = 0; kt < 8; ++kt) {
    v16bf a = load_afrag(Gs, strip, lane, kt*32);
    #pragma unroll
    for (int i = 0; i < 8; ++i) {
      v16bf bf = load_bfrag(wop, kt*16 + half*8 + i, lane);
      acc[i] = wmma_bf16(a, bf, acc[i]);
    }
  }
  #pragma unroll
  for (int i = 0; i < 8; ++i) {
    int col = (half*8 + i)*16 + (lane & 15);
    float obc = ob[col];
    #pragma unroll
    for (int r = 0; r < 8; ++r) {
      size_t rg = r0 + strip*16 + r + mhi;
      out[rg*NOUT + col] = acc[i][r] + obc;
    }
  }
}

// ---------------- launcher ----------------
extern "C" void kernel_launch(void* const* d_in, const int* in_sizes, int n_in,
                              void* d_out, int out_size, void* d_ws, size_t ws_size,
                              hipStream_t stream) {
  const float* q_data   = (const float*)d_in[0];
  const float* m_data   = (const float*)d_in[1];
  const float* q_mask   = (const float*)d_in[2];
  // d_in[3] = bias (ignored by module)
  const float* query_w  = (const float*)d_in[4];
  const float* key_w    = (const float*)d_in[5];
  const float* value_w  = (const float*)d_in[6];
  const float* gating_w = (const float*)d_in[7];
  const float* gating_b = (const float*)d_in[8];
  const float* output_w = (const float*)d_in[9];
  const float* output_b = (const float*)d_in[10];
  float* out = (float*)d_out;

  float* ws   = (float*)d_ws;
  float* qavg = ws;                         // 128*256
  float* qh   = ws + 32768;                 // 128*256
  float* wa   = ws + 65536;                 // 128*256
  float* kbuf = ws + 98304;                 // 262144*32
  float* vbuf = kbuf + (size_t)NB*NN*NKD;   // 262144*32
  bf16*  wgp  = (bf16*)(vbuf + (size_t)NB*NN*NKD);
  bf16*  wop  = wgp + 65536;
  bf16*  wkvp = wop + 65536;

  const int RT = NB*NN/64;                  // 4096 row tiles

  pack_wmma_b<<<288, 32, 0, stream>>>(gating_w, output_w, key_w, value_w, wgp, wop, wkvp);
  pool_query<<<NB, 256, 0, stream>>>(q_data, q_mask, qavg);
  project_q<<<NB, 256, 0, stream>>>(qavg, query_w, qh);
  kv_proj_kernel<<<RT, 256, (size_t)64*LDA*sizeof(bf16), stream>>>(m_data, wkvp, kbuf, vbuf);
  attention_pool<<<NB, 256, (2*256*33 + 8*256 + 256)*sizeof(float), stream>>>(
      kbuf, vbuf, qh, q_mask, wa);
  gate_output_kernel<<<RT, 256, (size_t)2*64*LDA*sizeof(bf16), stream>>>(
      q_data, wgp, wop, gating_b, wa, output_b, out);
}
